// TGN_20349555048573
// MI455X (gfx1250) — compile-verified, matrix operand order checked by reading, not run on hardware
//
#include <hip/hip_runtime.h>
#include <hip/hip_bf16.h>
#include <math.h>

typedef _Float16 h16v __attribute__((ext_vector_type(16)));
typedef _Float16 h8v  __attribute__((ext_vector_type(8)));
typedef float    f8v  __attribute__((ext_vector_type(8)));

#define N_NODES 200000
#define BATCH   4096
#define KNBR    20
#define NG      5      // neighbors per group
#define NGROUP  4      // KNBR / NG
#define NMSG    100000
#define DIM     128    // D = DT = DM = MSG
#define RAWD    512
#define HIDD    256
#define QDIM    256
#define KDIM    384

// f16 weight pool offsets (in halfs)
#define OFF_W1  0         // 256x512
#define OFF_W2  131072    // 128x256
#define OFF_WI  163840    // 384x128
#define OFF_WH  212992    // 384x128
#define OFF_WQ  262144    // 256x256
#define OFF_WK  327680    // 256x384
#define OFF_WV  425984    // 256x384
#define OFF_WO  524288    // 256x256
#define OFF_MW1 589824    // 128x384
#define OFF_MW2 638976    // 128x128

// workspace byte offsets
#define WS_LASTPOS 0
#define WS_MEMUPD  800000ull
#define WS_WPOOL   103200000ull

__device__ __forceinline__ f8v wmma32(h16v a, h16v b, f8v c) {
  return __builtin_amdgcn_wmma_f32_16x16x32_f16(false, a, false, b, (short)0, c,
                                                false, false);
}

// A fragment from row-major f16 LDS tile (16 rows, strideH halfs, strideH%8==0)
__device__ __forceinline__ h16v ldsA(const _Float16* base, int strideH, int kk, int lane) {
  int m  = lane & 15;
  int hb = (lane >> 4) << 3;
  const _Float16* p = base + m * strideH + kk + hb;
  h8v lo = *(const h8v*)p;
  h8v hi = *(const h8v*)(p + 16);
  h16v r;
#pragma unroll
  for (int i = 0; i < 8; ++i) { r[i] = lo[i]; r[i + 8] = hi[i]; }
  return r;
}

// B fragment: weight row-major [Nout][Kin] f16, B[k][n] = W[nbase+n][k]
__device__ __forceinline__ h16v ldB(const _Float16* __restrict__ W, int Kin,
                                    int nbase, int kk, int lane) {
  int n  = nbase + (lane & 15);
  int hb = (lane >> 4) << 3;
  const _Float16* p = W + (size_t)n * Kin + kk + hb;
  h8v lo = *(const h8v*)p;
  h8v hi = *(const h8v*)(p + 16);
  h16v r;
#pragma unroll
  for (int i = 0; i < 8; ++i) { r[i] = lo[i]; r[i + 8] = hi[i]; }
  return r;
}

// 8-element f32->f16 chunk helpers (dst 16B-aligned LDS)
__device__ __forceinline__ h8v cvt8(float4 a, float4 b) {
  h8v v;
  v[0] = (_Float16)a.x; v[1] = (_Float16)a.y; v[2] = (_Float16)a.z; v[3] = (_Float16)a.w;
  v[4] = (_Float16)b.x; v[5] = (_Float16)b.y; v[6] = (_Float16)b.z; v[7] = (_Float16)b.w;
  return v;
}
__device__ __forceinline__ void store_h8(_Float16* dst, const float* src) {
  const float4* p = (const float4*)src;
  *(h8v*)dst = cvt8(p[0], p[1]);
}
__device__ __forceinline__ h8v sum8(const float* a, const float* b) {
  const float4* pa = (const float4*)a; const float4* pb = (const float4*)b;
  float4 x0 = pa[0], x1 = pa[1], y0 = pb[0], y1 = pb[1];
  float4 s0 = make_float4(x0.x + y0.x, x0.y + y0.y, x0.z + y0.z, x0.w + y0.w);
  float4 s1 = make_float4(x1.x + y1.x, x1.y + y1.y, x1.z + y1.z, x1.w + y1.w);
  return cvt8(s0, s1);
}
__device__ __forceinline__ void store_cos8(_Float16* dst, float dt,
                                           const float* tw, const float* tb, int off) {
  h8v v;
#pragma unroll
  for (int i = 0; i < 8; ++i)
    v[i] = (_Float16)__cosf(dt * tw[off + i] + tb[off + i]);
  *(h8v*)dst = v;
}

// ---------------------------------------------------------------- utility
__global__ void k_init_lastpos(int* __restrict__ lastpos) {
  int i = blockIdx.x * blockDim.x + threadIdx.x;
  if (i < N_NODES) lastpos[i] = -1;
}
__global__ void k_scatter_max(const int* __restrict__ msg_src,
                              int* __restrict__ lastpos) {
  int i = blockIdx.x * blockDim.x + threadIdx.x;
  if (i < NMSG) atomicMax(&lastpos[msg_src[i]], i);
}
__global__ void k_cvt(const float* __restrict__ src, _Float16* __restrict__ dst, int n) {
  int i = blockIdx.x * blockDim.x + threadIdx.x;
  if (i < n) dst[i] = (_Float16)src[i];
}

// ---------------------------------------------------------------- kernel 2
// one wave, 4 M-tiles (64 nodes): message MLP + GRU -> mem_upd
// N-tile loop outermost so every weight fragment is reused 4x.
__global__ __launch_bounds__(32) void k_mem_update(
    const int* __restrict__ msg_src, const int* __restrict__ msg_dst,
    const int* __restrict__ msg_eidx, const int* __restrict__ msg_t,
    const int* __restrict__ last_update, const float* __restrict__ memory,
    const float* __restrict__ edge_feats,
    const float* __restrict__ time_w, const float* __restrict__ time_b,
    const float* __restrict__ msg_b1, const float* __restrict__ msg_b2,
    const float* __restrict__ gru_bi, const float* __restrict__ gru_bh,
    const int* __restrict__ lastpos, const _Float16* __restrict__ wpool,
    float* __restrict__ memupd)
{
  __shared__ int   s_s[64], s_d[64], s_e[64], s_has[64];
  __shared__ float s_dt[64];
  __shared__ __align__(16) _Float16 s_raw[4 * 16 * 520];  // 66560 B
  __shared__ __align__(16) _Float16 s_h1 [4 * 16 * 264];  // 33792 B
  __shared__ __align__(16) _Float16 s_msg[4 * 16 * 136];  // 17408 B
  __shared__ __align__(16) _Float16 s_mem[4 * 16 * 136];  // 17408 B

  int lane = threadIdx.x;
  int base = blockIdx.x * 64;
  int nl = lane & 15, hh = lane >> 4;

  for (int r = lane; r < 64; r += 32) {
    int node = base + r;
    int lp = lastpos[node];
    int has = (lp >= 0);
    int idx = has ? lp : 0;
    int s = msg_src[idx];
    s_s[r] = s; s_d[r] = msg_dst[idx]; s_e[r] = msg_eidx[idx];
    s_dt[r] = (float)(msg_t[idx] - last_update[s]);
    s_has[r] = has;
  }
  __syncthreads();

  // build raw tiles [4][16][512] f16 (stride 520), 8-elem chunks
  for (int i = lane; i < 4096; i += 32) {
    int t = i >> 10, rem = i & 1023;
    int r = rem >> 6, c = (rem & 63) << 3;
    int row = t * 16 + r;
    _Float16* dst = s_raw + t * 8320 + r * 520 + c;
    int region = c >> 7, off = c & 127;
    if (region == 0)      store_h8(dst, memory     + (size_t)s_s[row] * DIM + off);
    else if (region == 1) store_h8(dst, memory     + (size_t)s_d[row] * DIM + off);
    else if (region == 2) store_h8(dst, edge_feats + (size_t)s_e[row] * DIM + off);
    else                  store_cos8(dst, s_dt[row], time_w, time_b, off);
  }
  // stage memory rows of the 64 nodes as f16 (for gh GEMM)
  for (int i = lane; i < 1024; i += 32) {
    int t = i >> 8, rem = i & 255;
    int r = rem >> 4, c = (rem & 15) << 3;
    store_h8(s_mem + t * 2176 + r * 136 + c,
             memory + (size_t)(base + t * 16 + r) * DIM + c);
  }
  __syncthreads();

  // ---- GEMM1: raw @ W1^T -> h1 (B fragment reused across 4 tiles)
  for (int nt = 0; nt < 16; ++nt) {
    f8v acc[4] = {};
    for (int kk = 0; kk < RAWD; kk += 32) {
      h16v b = ldB(wpool + OFF_W1, RAWD, nt * 16, kk, lane);
#pragma unroll
      for (int t = 0; t < 4; ++t)
        acc[t] = wmma32(ldsA(s_raw + t * 8320, 520, kk, lane), b, acc[t]);
    }
    float bias = msg_b1[nt * 16 + nl];
#pragma unroll
    for (int t = 0; t < 4; ++t)
#pragma unroll
      for (int j = 0; j < 8; ++j)
        s_h1[t * 4224 + (j + 8 * hh) * 264 + nt * 16 + nl] =
            (_Float16)fmaxf(acc[t][j] + bias, 0.f);
  }
  __syncthreads();

  // ---- GEMM2: h1 @ W2^T -> msg
  for (int nt = 0; nt < 8; ++nt) {
    f8v acc[4] = {};
    for (int kk = 0; kk < HIDD; kk += 32) {
      h16v b = ldB(wpool + OFF_W2, HIDD, nt * 16, kk, lane);
#pragma unroll
      for (int t = 0; t < 4; ++t)
        acc[t] = wmma32(ldsA(s_h1 + t * 4224, 264, kk, lane), b, acc[t]);
    }
    float bias = msg_b2[nt * 16 + nl];
#pragma unroll
    for (int t = 0; t < 4; ++t)
#pragma unroll
      for (int j = 0; j < 8; ++j)
        s_msg[t * 2176 + (j + 8 * hh) * 136 + nt * 16 + nl] =
            (_Float16)(acc[t][j] + bias);
  }
  __syncthreads();

  // ---- GRU gates, output column tile at a time
  for (int ct = 0; ct < 8; ++ct) {
    f8v ai[4][3] = {};
    f8v ah[4][3] = {};
#pragma unroll
    for (int g = 0; g < 3; ++g) {
      for (int kk = 0; kk < DIM; kk += 32) {
        h16v bi = ldB(wpool + OFF_WI, DIM, g * 128 + ct * 16, kk, lane);
        h16v bh = ldB(wpool + OFF_WH, DIM, g * 128 + ct * 16, kk, lane);
#pragma unroll
        for (int t = 0; t < 4; ++t) {
          ai[t][g] = wmma32(ldsA(s_msg + t * 2176, 136, kk, lane), bi, ai[t][g]);
          ah[t][g] = wmma32(ldsA(s_mem + t * 2176, 136, kk, lane), bh, ah[t][g]);
        }
      }
    }
    int c = ct * 16 + nl;
    float bir = gru_bi[c],       bhr = gru_bh[c];
    float biz = gru_bi[128 + c], bhz = gru_bh[128 + c];
    float bin = gru_bi[256 + c], bhn = gru_bh[256 + c];
#pragma unroll
    for (int t = 0; t < 4; ++t) {
#pragma unroll
      for (int j = 0; j < 8; ++j) {
        int mm = j + 8 * hh;
        size_t node = (size_t)(base + t * 16 + mm);
        float old = memory[node * DIM + c];
        float r = 1.f / (1.f + __expf(-(ai[t][0][j] + bir + ah[t][0][j] + bhr)));
        float z = 1.f / (1.f + __expf(-(ai[t][1][j] + biz + ah[t][1][j] + bhz)));
        float nn = tanhf(ai[t][2][j] + bin + r * (ah[t][2][j] + bhn));
        float out = s_has[t * 16 + mm] ? ((1.f - z) * nn + z * old) : old;
        memupd[node * DIM + c] = out;
      }
    }
  }
}

// ---------------------------------------------------------------- kernel 3
// 8 waves (256 threads) per 16 query rows; neighbors in 4 groups of 5
// so Wk/Wv fragments are reused across 5 neighbor A-tiles.
__global__ __launch_bounds__(256) void k_embed(
    const int* __restrict__ src_nodes, const int* __restrict__ dst_nodes,
    const int* __restrict__ neg_nodes, const int* __restrict__ edge_times,
    const int* __restrict__ neighbors, const int* __restrict__ nbr_eidx,
    const int* __restrict__ nbr_et,
    const float* __restrict__ node_feats, const float* __restrict__ edge_feats,
    const float* __restrict__ time_w, const float* __restrict__ time_b,
    const float* __restrict__ bq, const float* __restrict__ bk,
    const float* __restrict__ bv, const float* __restrict__ bo,
    const float* __restrict__ mb1, const float* __restrict__ mb2,
    const float* __restrict__ memupd, const _Float16* __restrict__ wpool,
    float* __restrict__ out)
{
  __shared__ __align__(16) _Float16 s_kin[NG * 16 * 392];  // 62720 B
  __shared__ __align__(16) float    s_q[16 * 260];         // q, then ctx
  __shared__ __align__(16) _Float16 s_stage[16 * 264];     // qin / ctx f16 / hbuf
  __shared__ float s_attn[16 * 2 * KNBR];
  __shared__ __align__(16) _Float16 s_srcf[16 * 136];
  __shared__ int s_node[16], s_allm[16];
  __shared__ int s_nb[NG * 16], s_ne[NG * 16], s_padall[KNBR * 16];
  __shared__ float s_ts[16], s_dtn[NG * 16];

  int tid = threadIdx.x;
  int lane = tid & 31;
  int w = tid >> 5;                 // wave id 0..7
  int nl = lane & 15, hh = lane >> 4;
  int base = blockIdx.x * 16;

  if (tid < 16) {
    int q = base + tid;
    int which = q >> 12;            // / BATCH
    int bi = q & (BATCH - 1);
    int node = (which == 0) ? src_nodes[bi]
             : (which == 1) ? dst_nodes[bi] : neg_nodes[bi];
    s_node[tid] = node;
    s_ts[tid] = (float)edge_times[bi];
    s_allm[tid] = 1;
  }
  for (int i = tid; i < 16 * 2 * KNBR; i += 256) s_attn[i] = 0.f;
  __syncthreads();

  // ---- build q_in (f16 in s_stage, stride 264) + srcfeat
  for (int i = tid; i < 512; i += 256) {          // 16 rows x 32 chunks
    int r = i >> 5, c = (i & 31) << 3;
    if (c < 128) {
      int nd = s_node[r];
      h8v v = sum8(memupd + (size_t)nd * DIM + c, node_feats + (size_t)nd * DIM + c);
      *(h8v*)(s_stage + r * 264 + c) = v;
      *(h8v*)(s_srcf + r * 136 + c) = v;
    } else {
      store_cos8(s_stage + r * 264 + c, 0.f, time_w, time_b, c - 128);
    }
  }
  __syncthreads();

  // ---- q GEMM (wave w owns nt = 2w, 2w+1)
#pragma unroll
  for (int tt = 0; tt < 2; ++tt) {
    int nt = w * 2 + tt;
    f8v acc = {};
    for (int kk = 0; kk < QDIM; kk += 32)
      acc = wmma32(ldsA(s_stage, 264, kk, lane),
                   ldB(wpool + OFF_WQ, QDIM, nt * 16, kk, lane), acc);
    float bias = bq[nt * 16 + nl];
#pragma unroll
    for (int jj = 0; jj < 8; ++jj)
      s_q[(jj + 8 * hh) * 260 + nt * 16 + nl] = acc[jj] + bias;
  }
  __syncthreads();

  const float SCALE = 0.08838834764831843f;       // 1/sqrt(128)

  // ---- pass 1: keys + score partials (online fold from fragments)
  for (int g = 0; g < NGROUP; ++g) {
    int jbase = g * NG;
    // group metadata
    if (tid < NG * 16) {
      int jm = tid >> 4, r = tid & 15, j = jbase + jm;
      size_t qn = (size_t)(base + r);
      int nb = neighbors[qn * KNBR + j];
      s_nb[jm * 16 + r] = nb;
      s_padall[j * 16 + r] = (nb == 0);
      if (nb != 0) s_allm[r] = 0;
      s_ne[jm * 16 + r] = nbr_eidx[qn * KNBR + j];
      s_dtn[jm * 16 + r] = s_ts[r] - (float)nbr_et[qn * KNBR + j];
    }
    __syncthreads();
    // build 5 k_in tiles
    for (int i = tid; i < NG * 16 * 48; i += 256) {
      int jm = i / 768, rem = i - jm * 768;
      int r = rem / 48, c = (rem - r * 48) << 3;
      _Float16* dst = s_kin + jm * 6272 + r * 392 + c;
      if (c < 128) {
        int nd = s_nb[jm * 16 + r];
        *(h8v*)dst = sum8(memupd + (size_t)nd * DIM + c,
                          node_feats + (size_t)nd * DIM + c);
      } else if (c < 256) {
        store_cos8(dst, s_dtn[jm * 16 + r], time_w, time_b, c - 128);
      } else {
        store_h8(dst, edge_feats + (size_t)s_ne[jm * 16 + r] * DIM + (c - 256));
      }
    }
    __syncthreads();

    int head = w >> 2;              // n range per wave is head-uniform
    float part[NG][8] = {};
#pragma unroll
    for (int tt = 0; tt < 2; ++tt) {
      int nt = w * 2 + tt;
      f8v acc5[NG] = {};
      for (int kk = 0; kk < KDIM; kk += 32) {
        h16v b = ldB(wpool + OFF_WK, KDIM, nt * 16, kk, lane);
#pragma unroll
        for (int jm = 0; jm < NG; ++jm)
          acc5[jm] = wmma32(ldsA(s_kin + jm * 6272, 392, kk, lane), b, acc5[jm]);
      }
      int n = nt * 16 + nl;
      float bias = bk[n];
#pragma unroll
      for (int jj = 0; jj < 8; ++jj) {
        float qv = s_q[(jj + 8 * hh) * 260 + n];
#pragma unroll
        for (int jm = 0; jm < NG; ++jm)
          part[jm][jj] += (acc5[jm][jj] + bias) * qv;
      }
    }
#pragma unroll
    for (int jj = 0; jj < 8; ++jj) {
      int mm = jj + 8 * hh;
#pragma unroll
      for (int jm = 0; jm < NG; ++jm)
        atomicAdd(&s_attn[(mm * 2 + head) * KNBR + jbase + jm], part[jm][jj]);
    }
    __syncthreads();
  }

  // ---- softmax per (row, head) with pad mask
  if (tid < 32) {
    int mm = tid & 15, hd = tid >> 4;
    float* sp = s_attn + (mm * 2 + hd) * KNBR;
    float mx = -1e30f;
    for (int j = 0; j < KNBR; ++j) {
      float s = sp[j] * SCALE;
      if (s_padall[j * 16 + mm]) s = -1e9f;
      sp[j] = s;
      mx = fmaxf(mx, s);
    }
    float sum = 0.f;
    for (int j = 0; j < KNBR; ++j) { float e = __expf(sp[j] - mx); sp[j] = e; sum += e; }
    float inv = 1.f / sum;
    for (int j = 0; j < KNBR; ++j) sp[j] *= inv;
  }
  __syncthreads();

  // ---- pass 2: values, weighted context accumulated in s_q
  for (int i = tid; i < 16 * 260; i += 256) s_q[i] = 0.f;
  __syncthreads();
  for (int g = 0; g < NGROUP; ++g) {
    int jbase = g * NG;
    if (tid < NG * 16) {
      int jm = tid >> 4, r = tid & 15, j = jbase + jm;
      size_t qn = (size_t)(base + r);
      int nb = neighbors[qn * KNBR + j];
      s_nb[jm * 16 + r] = nb;
      s_ne[jm * 16 + r] = nbr_eidx[qn * KNBR + j];
      s_dtn[jm * 16 + r] = s_ts[r] - (float)nbr_et[qn * KNBR + j];
    }
    __syncthreads();
    for (int i = tid; i < NG * 16 * 48; i += 256) {
      int jm = i / 768, rem = i - jm * 768;
      int r = rem / 48, c = (rem - r * 48) << 3;
      _Float16* dst = s_kin + jm * 6272 + r * 392 + c;
      if (c < 128) {
        int nd = s_nb[jm * 16 + r];
        *(h8v*)dst = sum8(memupd + (size_t)nd * DIM + c,
                          node_feats + (size_t)nd * DIM + c);
      } else if (c < 256) {
        store_cos8(dst, s_dtn[jm * 16 + r], time_w, time_b, c - 128);
      } else {
        store_h8(dst, edge_feats + (size_t)s_ne[jm * 16 + r] * DIM + (c - 256));
      }
    }
    __syncthreads();

    int head = w >> 2;
#pragma unroll
    for (int tt = 0; tt < 2; ++tt) {
      int nt = w * 2 + tt;
      f8v acc5[NG] = {};
      for (int kk = 0; kk < KDIM; kk += 32) {
        h16v b = ldB(wpool + OFF_WV, KDIM, nt * 16, kk, lane);
#pragma unroll
        for (int jm = 0; jm < NG; ++jm)
          acc5[jm] = wmma32(ldsA(s_kin + jm * 6272, 392, kk, lane), b, acc5[jm]);
      }
      int n = nt * 16 + nl;
      float bias = bv[n];
#pragma unroll
      for (int jj = 0; jj < 8; ++jj) {
        int mm = jj + 8 * hh;
        float* cptr = &s_q[mm * 260 + n];          // unique (mm,n) per lane
        float acc = *cptr;
#pragma unroll
        for (int jm = 0; jm < NG; ++jm) {
          float wv = s_attn[(mm * 2 + head) * KNBR + jbase + jm];
          acc += wv * (acc5[jm][jj] + bias);
        }
        *cptr = acc;
      }
    }
    __syncthreads();
  }

  // ---- ctx -> f16 staging
  for (int i = tid; i < 512; i += 256) {
    int r = i >> 5, c = (i & 31) << 3;
    const float4* p = (const float4*)(s_q + r * 260 + c);
    *(h8v*)(s_stage + r * 264 + c) = cvt8(p[0], p[1]);
  }
  __syncthreads();

  // ---- attn_out = ctx @ Wo^T + bo ; zero all-masked ; into merge cols 0..255
#pragma unroll
  for (int tt = 0; tt < 2; ++tt) {
    int nt = w * 2 + tt;
    f8v acc = {};
    for (int kk = 0; kk < QDIM; kk += 32)
      acc = wmma32(ldsA(s_stage, 264, kk, lane),
                   ldB(wpool + OFF_WO, QDIM, nt * 16, kk, lane), acc);
    float bias = bo[nt * 16 + nl];
#pragma unroll
    for (int jj = 0; jj < 8; ++jj) {
      int mm = jj + 8 * hh;
      float v = s_allm[mm] ? 0.f : (acc[jj] + bias);
      s_kin[mm * 392 + nt * 16 + nl] = (_Float16)v;
    }
  }
  __syncthreads();
  for (int i = tid; i < 256; i += 256) {          // merge cols 256..383
    int r = i >> 4, c = (i & 15) << 3;
    *(h8v*)(s_kin + r * 392 + 256 + c) = *(const h8v*)(s_srcf + r * 136 + c);
  }
  __syncthreads();

  // ---- merge GEMM1 (relu) -> hbuf in s_stage
  {
    int nt = w;                                    // 8 waves, 8 n-tiles
    f8v acc = {};
    for (int kk = 0; kk < KDIM; kk += 32)
      acc = wmma32(ldsA(s_kin, 392, kk, lane),
                   ldB(wpool + OFF_MW1, KDIM, nt * 16, kk, lane), acc);
    float bias = mb1[nt * 16 + nl];
#pragma unroll
    for (int jj = 0; jj < 8; ++jj)
      s_stage[(jj + 8 * hh) * 136 + nt * 16 + nl] =
          (_Float16)fmaxf(acc[jj] + bias, 0.f);
  }
  __syncthreads();

  // ---- merge GEMM2 -> out
  {
    int nt = w;
    f8v acc = {};
    for (int kk = 0; kk < DIM; kk += 32)
      acc = wmma32(ldsA(s_stage, 136, kk, lane),
                   ldB(wpool + OFF_MW2, DIM, nt * 16, kk, lane), acc);
    float bias = mb2[nt * 16 + nl];
#pragma unroll
    for (int jj = 0; jj < 8; ++jj) {
      int mm = jj + 8 * hh;
      out[(size_t)(base + mm) * DIM + nt * 16 + nl] = acc[jj] + bias;
    }
  }
}

// ---------------------------------------------------------------- launch
extern "C" void kernel_launch(void* const* d_in, const int* in_sizes, int n_in,
                              void* d_out, int out_size, void* d_ws, size_t ws_size,
                              hipStream_t stream) {
  (void)in_sizes; (void)n_in; (void)out_size; (void)ws_size;

  const int*   src_nodes   = (const int*)d_in[0];
  const int*   dst_nodes   = (const int*)d_in[1];
  const int*   neg_nodes   = (const int*)d_in[2];
  const int*   edge_times  = (const int*)d_in[3];
  const int*   neighbors   = (const int*)d_in[4];
  const int*   nbr_eidx    = (const int*)d_in[5];
  const int*   nbr_et      = (const int*)d_in[6];
  const int*   msg_src     = (const int*)d_in[7];
  const int*   msg_dst     = (const int*)d_in[8];
  const int*   msg_eidx    = (const int*)d_in[9];
  const int*   msg_t       = (const int*)d_in[10];
  const int*   last_update = (const int*)d_in[11];
  const float* node_feats  = (const float*)d_in[12];
  const float* edge_feats  = (const float*)d_in[13];
  const float* memory      = (const float*)d_in[14];
  const float* time_w      = (const float*)d_in[15];
  const float* time_b      = (const float*)d_in[16];
  const float* msg_W1      = (const float*)d_in[17];
  const float* msg_b1      = (const float*)d_in[18];
  const float* msg_W2      = (const float*)d_in[19];
  const float* msg_b2      = (const float*)d_in[20];
  const float* gru_Wi      = (const float*)d_in[21];
  const float* gru_Wh      = (const float*)d_in[22];
  const float* gru_bi      = (const float*)d_in[23];
  const float* gru_bh      = (const float*)d_in[24];
  const float* Wq          = (const float*)d_in[25];
  const float* bq          = (const float*)d_in[26];
  const float* Wk          = (const float*)d_in[27];
  const float* bk          = (const float*)d_in[28];
  const float* Wv          = (const float*)d_in[29];
  const float* bv          = (const float*)d_in[30];
  const float* Wo          = (const float*)d_in[31];
  const float* bo          = (const float*)d_in[32];
  const float* mW1         = (const float*)d_in[33];
  const float* mb1         = (const float*)d_in[34];
  const float* mW2         = (const float*)d_in[35];
  const float* mb2         = (const float*)d_in[36];

  char* ws = (char*)d_ws;
  int*      lastpos = (int*)(ws + WS_LASTPOS);
  float*    memupd  = (float*)(ws + WS_MEMUPD);
  _Float16* wpool   = (_Float16*)(ws + WS_WPOOL);
  float*    out     = (float*)d_out;

  k_init_lastpos<<<(N_NODES + 255) / 256, 256, 0, stream>>>(lastpos);
  k_scatter_max<<<(NMSG + 255) / 256, 256, 0, stream>>>(msg_src, lastpos);

  struct { const float* s; int off; int n; } cv[10] = {
    { msg_W1, OFF_W1,  HIDD * RAWD }, { msg_W2, OFF_W2,  DIM * HIDD },
    { gru_Wi, OFF_WI,  3 * DIM * DIM }, { gru_Wh, OFF_WH, 3 * DIM * DIM },
    { Wq,     OFF_WQ,  QDIM * QDIM }, { Wk,     OFF_WK,  QDIM * KDIM },
    { Wv,     OFF_WV,  QDIM * KDIM }, { Wo,     OFF_WO,  QDIM * QDIM },
    { mW1,    OFF_MW1, DIM * KDIM },  { mW2,    OFF_MW2, DIM * DIM },
  };
  for (int i = 0; i < 10; ++i)
    k_cvt<<<(cv[i].n + 255) / 256, 256, 0, stream>>>(cv[i].s, wpool + cv[i].off, cv[i].n);

  k_mem_update<<<N_NODES / 64, 32, 0, stream>>>(
      msg_src, msg_dst, msg_eidx, msg_t, last_update, memory, edge_feats,
      time_w, time_b, msg_b1, msg_b2, gru_bi, gru_bh, lastpos, wpool, memupd);

  k_embed<<<(3 * BATCH) / 16, 256, 0, stream>>>(
      src_nodes, dst_nodes, neg_nodes, edge_times, neighbors, nbr_eidx, nbr_et,
      node_feats, edge_feats, time_w, time_b, bq, bk, bv, bo, mb1, mb2,
      memupd, wpool, out);
}
